// ModularSSMWorldModel_5257039970375
// MI455X (gfx1250) — matrix operand order repaired and checked by reading, not compiled, not run on hardware
//
#include <hip/hip_runtime.h>

// ---------------------------------------------------------------------------
// Problem constants (match reference)
// ---------------------------------------------------------------------------
#define INPUT_DIM 256
#define LATENT    1024
#define MC        16
#define HID       128
#define BATCH     32
#define TSTEPS    1024
#define BW        64      // slice width, modules 0..14
#define STEP      48      // slice step
#define S_LAST    720
#define W_LAST    304     // last slice width
#define KA_MOD    320     // A width for modules 0..14 (256 + 64)
#define KA_LAST   576     // padded A width for last module (256+304 -> 576)
#define KT1_MOD   10      // 320/32
#define KT1_LAST  18      // 576/32
#define NT2_MOD   4       // 64/16
#define NT2_LAST  19      // 304/16
#define UPD_STRIDE 40448  // floats per private-update buffer: 15*32*64 + 32*304
#define UPD_LAST_OFF 30720

typedef __attribute__((ext_vector_type(16))) __bf16 v16bf;
typedef __attribute__((ext_vector_type(8)))  __bf16 v8bf;
typedef __attribute__((ext_vector_type(8)))  float  v8f;

union BF16x16 { v16bf v; v8bf h[2]; };

__device__ __forceinline__ unsigned short f2bf(float f) {
    unsigned int u = __float_as_uint(f);
    return (unsigned short)((u + 0x7FFFu + ((u >> 16) & 1u)) >> 16);
}

__device__ __forceinline__ v8f wmma_bf16(v16bf a, v16bf b, v8f c) {
    return __builtin_amdgcn_wmma_f32_16x16x32_bf16(
        /*neg_a=*/false, a, /*neg_b=*/false, b,
        /*c_mod=*/(short)0, c, /*reuse_a=*/false, /*reuse_b=*/false);
}

// Async DMA: 16 bytes global -> LDS, tracked on ASYNCcnt (no VGPR round-trip).
__device__ __forceinline__ void async_copy16(unsigned lds_off, const void* gsrc) {
    asm volatile("global_load_async_to_lds_b128 %0, %1, off"
                 :: "v"(lds_off), "v"(gsrc) : "memory");
}
__device__ __forceinline__ void wait_async0() {
    asm volatile("s_wait_asynccnt 0x0" ::: "memory");
}

// A-fragment, contiguous-K order matching the B packing. Lane holds M=lane%16; K values
// {kb..kb+15} with kb = ktile*32 + (lane>=16 ? 16 : 0).
__device__ __forceinline__ v16bf lds_afrag(const unsigned short* rowPtr, int kb) {
    BF16x16 u;
    u.h[0] = *(const v8bf*)(rowPtr + kb);
    u.h[1] = *(const v8bf*)(rowPtr + kb + 8);
    return u.v;
}

// B-fragment from pre-packed global weights: lane's 16 values contiguous.
__device__ __forceinline__ v16bf ldg_bfrag(const unsigned short* p) {
    BF16x16 u;
    u.h[0] = *(const v8bf*)(p);
    u.h[1] = *(const v8bf*)(p + 8);
    return u.v;
}

// ---------------------------------------------------------------------------
// Prep kernel 1: inputs (B,T,C) fp32 -> xbf (T,B,C) bf16
// ---------------------------------------------------------------------------
__global__ void k_prep_x(const float* __restrict__ in, unsigned short* __restrict__ xbf) {
    const int stride = gridDim.x * blockDim.x;
    for (int p = blockIdx.x * blockDim.x + threadIdx.x;
         p < TSTEPS * BATCH * INPUT_DIM; p += stride) {
        int t = p >> 13;            // 8192 per timestep
        int r = p & 8191;
        int b = r >> 8;
        int c = r & 255;
        xbf[p] = f2bf(in[((size_t)b * TSTEPS + t) * INPUT_DIM + c]);
    }
}

// ---------------------------------------------------------------------------
// Prep kernel 2: pack all weights into bf16 WMMA B-fragments.
// B-frag layout (32x16, K x N): lane l holds N = l&15, K = kt*32 + (l&16) + e,
// e = 0..15 contiguous per lane. packed[frag*512 + l*16 + e].
// ---------------------------------------------------------------------------
__global__ void k_pack(const float* __restrict__ W1,  const float* __restrict__ W2,
                       const float* __restrict__ W1L, const float* __restrict__ W2L,
                       unsigned short* __restrict__ p1,  unsigned short* __restrict__ p2,
                       unsigned short* __restrict__ p1L, unsigned short* __restrict__ p2L) {
    const int stride = gridDim.x * blockDim.x;
    const int g0 = blockIdx.x * blockDim.x + threadIdx.x;

    // W1: (15, 320, 128) -> frags f = (m*8+n)*10 + kt
    for (int p = g0; p < 15 * 8 * KT1_MOD * 512; p += stride) {
        int e = p & 15, l = (p >> 4) & 31, f = p >> 9;
        int kt = f % 10, rest = f / 10, n = rest & 7, m = rest >> 3;
        int kk = kt * 32 + (l & 16) + e;
        int c  = n * 16 + (l & 15);
        p1[p] = f2bf(W1[((size_t)m * KA_MOD + kk) * HID + c]);
    }
    // W2: (15, 128, 64) -> frags f = (m*4+n)*4 + kt
    for (int p = g0; p < 15 * 4 * 4 * 512; p += stride) {
        int e = p & 15, l = (p >> 4) & 31, f = p >> 9;
        int kt = f & 3, rest = f >> 2, n = rest & 3, m = rest >> 2;
        int kk = kt * 32 + (l & 16) + e;
        int c  = n * 16 + (l & 15);
        p2[p] = f2bf(W2[((size_t)m * HID + kk) * BW + c]);
    }
    // W1_last: (560, 128), K zero-padded to 576 -> frags f = n*18 + kt
    for (int p = g0; p < 8 * KT1_LAST * 512; p += stride) {
        int e = p & 15, l = (p >> 4) & 31, f = p >> 9;
        int kt = f % 18, n = f / 18;
        int kk = kt * 32 + (l & 16) + e;
        int c  = n * 16 + (l & 15);
        float v = (kk < INPUT_DIM + W_LAST) ? W1L[(size_t)kk * HID + c] : 0.0f;
        p1L[p] = f2bf(v);
    }
    // W2_last: (128, 304) -> frags f = n*4 + kt
    for (int p = g0; p < NT2_LAST * 4 * 512; p += stride) {
        int e = p & 15, l = (p >> 4) & 31, f = p >> 9;
        int kt = f & 3, n = f >> 2;
        int kk = kt * 32 + (l & 16) + e;
        int c  = n * 16 + (l & 15);
        p2L[p] = f2bf(W2L[(size_t)kk * W_LAST + c]);
    }
}

// ---------------------------------------------------------------------------
// Prep kernel 3: decay = sigmoid(decay_raw); inv_counts; zero state/barrier
// ---------------------------------------------------------------------------
__global__ void k_prep_misc(const float* __restrict__ decay_raw,
                            float* __restrict__ decay, float* __restrict__ invcnt,
                            float* __restrict__ stbuf, int* __restrict__ bar) {
    const int stride = gridDim.x * blockDim.x;
    const int g0 = blockIdx.x * blockDim.x + threadIdx.x;
    for (int j = g0; j < LATENT; j += stride) {
        decay[j] = 1.0f / (1.0f + __expf(-decay_raw[j]));
        int cnt = 0;
        #pragma unroll
        for (int i = 0; i < 15; ++i)
            if (j >= i * STEP && j < i * STEP + BW) cnt++;
        if (j >= S_LAST) cnt++;
        if (cnt == 0) cnt = 1;
        invcnt[j] = 1.0f / (float)cnt;
    }
    for (int j = g0; j < 2 * BATCH * LATENT; j += stride) stbuf[j] = 0.0f;
    for (int j = g0; j < 64; j += stride) bar[j] = 0;
}

// ---------------------------------------------------------------------------
// Grid barrier: cluster barrier (NOP if not launched as a cluster) + atomic
// sense-reversing barrier. Release fence on entry, acquire fence on exit,
// since the cross-block handoff now uses plain stores.
// ---------------------------------------------------------------------------
__device__ __forceinline__ void grid_barrier(int* cnt, volatile int* gen) {
    __threadfence();                         // release: flush this thread's stores
    __syncthreads();
    __builtin_amdgcn_s_cluster_barrier();    // real barrier when clustered; else S_NOP
    if (threadIdx.x == 0) {
        int g = *gen;
        if (atomicAdd(cnt, 1) == (int)gridDim.x - 1) {
            *cnt = 0;
            __threadfence();
            atomicAdd((int*)gen, 1);
        } else {
            while (*gen == g) { __builtin_amdgcn_s_sleep(2); }
        }
    }
    __syncthreads();
    __threadfence();                         // acquire: invalidate stale near caches
}

// ---------------------------------------------------------------------------
// Main persistent kernel: 16 workgroups (one per module), 256 threads (8 waves)
// One grid barrier per timestep; no atomics on the critical path.
// ---------------------------------------------------------------------------
__global__ void __launch_bounds__(256, 1)
k_scan(const unsigned short* __restrict__ xbf,
       const unsigned short* __restrict__ p1,  const unsigned short* __restrict__ p2,
       const unsigned short* __restrict__ p1L, const unsigned short* __restrict__ p2L,
       const float* __restrict__ b1,  const float* __restrict__ b2,
       const float* __restrict__ b1L, const float* __restrict__ b2L,
       const float* __restrict__ decay, const float* __restrict__ invcnt,
       float* __restrict__ stbuf, float* __restrict__ updbuf,
       int* __restrict__ barcnt, int* __restrict__ bargen,
       float* __restrict__ out) {
    __shared__ unsigned short Ast[BATCH][584];   // bf16 activations (A), 16B-aligned rows
    __shared__ unsigned short Hst[BATCH][136];   // bf16 hidden (silu output)

    const int tid  = threadIdx.x;
    const int lane = tid & 31;
    const int w    = tid >> 5;                    // wave id 0..7
    const int m    = blockIdx.x;                  // module id 0..15
    const bool last = (m == MC - 1);
    const int sOff = last ? S_LAST : m * STEP;
    const int slW  = last ? W_LAST : BW;
    const int KT1  = last ? KT1_LAST : KT1_MOD;
    const int NT2  = last ? NT2_LAST : NT2_MOD;
    const int nl   = lane & 15;
    const int rbase = (lane & 16) ? 8 : 0;        // C/D row offset for this lane half
    const int abase = (lane & 16) ? 16 : 0;       // A K-offset for this lane half

    const unsigned short* w1base =
        last ? (p1L + (size_t)(w * KT1_LAST) * 512)
             : (p1  + (size_t)((m * 8 + w) * KT1_MOD) * 512);

    const unsigned short* a0row = &Ast[nl][0];
    const unsigned short* a1row = &Ast[16 + nl][0];

    for (int t = 0; t < TSTEPS; ++t) {
        const float* cur = stbuf + (size_t)(t & 1) * (BATCH * LATENT);
        float*       nxt = stbuf + (size_t)((t + 1) & 1) * (BATCH * LATENT);
        float* ubuf = updbuf + (size_t)(t & 1) * UPD_STRIDE;

        // ---- build A tile in LDS: x part via async DMA (global -> LDS) ----
        const unsigned short* xt = xbf + (size_t)t * (BATCH * INPUT_DIM);
        for (int i = tid; i < (BATCH * INPUT_DIM) / 8; i += 256) {
            int b = i >> 5;                // 32 16B-chunks per row
            int c = (i & 31) * 8;
            async_copy16((unsigned)(unsigned long long)&Ast[b][c],
                         (const void*)(xt + b * INPUT_DIM + c));
        }
        // state slice (fp32 -> bf16) via VALU
        for (int i = tid; i < BATCH * slW; i += 256) {
            int b = i / slW, c = i - b * slW;
            Ast[b][INPUT_DIM + c] = f2bf(cur[b * LATENT + sOff + c]);
        }
        if (last) {
            for (int i = tid; i < BATCH * 16; i += 256)
                Ast[i >> 4][560 + (i & 15)] = 0;
        }
        // prefetch next step's activations while we compute this one
        if (t + 1 < TSTEPS)
            __builtin_prefetch(xt + (BATCH * INPUT_DIM) + tid * 32, 0, 0);
        wait_async0();
        __syncthreads();

        // ---- GEMM1: h = silu(A(32xKA) * W1(KAx128) + b1); wave w owns N-tile w
        v8f acc0 = {};  // M-tile 0 (rows 0..15)
        v8f acc1 = {};  // M-tile 1 (rows 16..31)
        for (int kt = 0; kt < KT1; ++kt) {
            v16bf bf = ldg_bfrag(w1base + (size_t)kt * 512 + lane * 16);
            int kb = kt * 32 + abase;
            v16bf a0 = lds_afrag(a0row, kb);
            v16bf a1 = lds_afrag(a1row, kb);
            acc0 = wmma_bf16(a0, bf, acc0);
            acc1 = wmma_bf16(a1, bf, acc1);
        }
        {
            const int ncol = w * 16 + nl;
            const float bb = last ? b1L[ncol] : b1[m * HID + ncol];
            #pragma unroll
            for (int r = 0; r < 8; ++r) {
                float z0 = acc0[r] + bb;
                Hst[rbase + r][ncol]      = f2bf(z0 / (1.0f + __expf(-z0)));
                float z1 = acc1[r] + bb;
                Hst[16 + rbase + r][ncol] = f2bf(z1 / (1.0f + __expf(-z1)));
            }
        }
        __syncthreads();

        // ---- GEMM2: module-private update = h(32x128)*W2(128xslW) + b2 ----
        float* myupd = ubuf + (last ? UPD_LAST_OFF : m * (BATCH * BW));
        for (int tileIdx = w; tileIdx < 2 * NT2; tileIdx += 8) {
            int mt = tileIdx / NT2, nt = tileIdx % NT2;
            v8f acc = {};
            const unsigned short* w2base =
                last ? (p2L + (size_t)(nt * 4) * 512)
                     : (p2  + (size_t)((m * 4 + nt) * 4) * 512);
            const unsigned short* arow = &Hst[mt * 16 + nl][0];
            #pragma unroll
            for (int kt = 0; kt < 4; ++kt) {
                v16bf bf = ldg_bfrag(w2base + (size_t)kt * 512 + lane * 16);
                v16bf af = lds_afrag(arow, kt * 32 + abase);
                acc = wmma_bf16(af, bf, acc);
            }
            int ncol = nt * 16 + nl;
            float bv = last ? b2L[ncol] : b2[m * BW + ncol];
            #pragma unroll
            for (int r = 0; r < 8; ++r) {
                int grow = mt * 16 + rbase + r;
                myupd[grow * slW + ncol] = acc[r] + bv;    // plain store, no atomics
            }
        }

        // ---- single sync point: all module contributions visible ----
        grid_barrier(barcnt, bargen);

        // ---- decay update for the columns THIS block reads next step.
        // Overlap columns are redundantly computed by both neighbors with
        // bitwise-identical inputs -> identical values (deterministic).
        for (int i = tid; i < BATCH * slW; i += 256) {
            int b = i / slW, c = i - b * slW;
            int j = sOff + c;
            float s = 0.0f;
            int i0 = j / STEP;                 // always covers j when i0 < 15
            if (i0 < 15) s += ubuf[i0 * (BATCH * BW) + b * BW + (j - i0 * STEP)];
            int i1 = i0 - 1;
            if (i1 >= 0 && (j - i1 * STEP) < BW)
                s += ubuf[i1 * (BATCH * BW) + b * BW + (j - i1 * STEP)];
            if (j >= S_LAST) s += ubuf[UPD_LAST_OFF + b * W_LAST + (j - S_LAST)];
            float d  = decay[j];
            float ns = d * cur[b * LATENT + j] + (1.0f - d) * (s * invcnt[j]);
            nxt[b * LATENT + j] = ns;
            out[(size_t)b * TSTEPS * LATENT + (size_t)t * LATENT + j] = ns;
            if (t == TSTEPS - 1)
                out[(size_t)BATCH * TSTEPS * LATENT + b * LATENT + j] = ns;
        }
        __threadfence_block();   // own writes to nxt visible block-locally
        __syncthreads();
    }
}

// ---------------------------------------------------------------------------
// Host-side launch
// ---------------------------------------------------------------------------
extern "C" void kernel_launch(void* const* d_in, const int* in_sizes, int n_in,
                              void* d_out, int out_size, void* d_ws, size_t ws_size,
                              hipStream_t stream) {
    const float* inputs    = (const float*)d_in[0];
    const float* W1        = (const float*)d_in[1];
    const float* b1        = (const float*)d_in[2];
    const float* W2        = (const float*)d_in[3];
    const float* b2        = (const float*)d_in[4];
    const float* W1L       = (const float*)d_in[5];
    const float* b1L       = (const float*)d_in[6];
    const float* W2L       = (const float*)d_in[7];
    const float* b2L       = (const float*)d_in[8];
    const float* decay_raw = (const float*)d_in[9];
    float* out = (float*)d_out;

    char* ws = (char*)d_ws;
    size_t oXbf = 0;
    size_t oW1  = oXbf + (size_t)TSTEPS * BATCH * INPUT_DIM * 2;   // 16,777,216
    size_t oW2  = oW1  + (size_t)15 * 8 * KT1_MOD * 512 * 2;       // +1,228,800
    size_t oW1L = oW2  + (size_t)15 * 4 * 4 * 512 * 2;             // +245,760
    size_t oW2L = oW1L + (size_t)8 * KT1_LAST * 512 * 2;           // +147,456
    size_t oDec = oW2L + (size_t)NT2_LAST * 4 * 512 * 2;           // +77,824
    size_t oInv = oDec + 4096;
    size_t oSt  = oInv + 4096;
    size_t oUpd = oSt  + (size_t)2 * BATCH * LATENT * 4;           // double-buffered state
    size_t oBar = oUpd + (size_t)2 * UPD_STRIDE * 4;               // double-buffered updates

    unsigned short* xbf = (unsigned short*)(ws + oXbf);
    unsigned short* p1  = (unsigned short*)(ws + oW1);
    unsigned short* p2  = (unsigned short*)(ws + oW2);
    unsigned short* p1L = (unsigned short*)(ws + oW1L);
    unsigned short* p2L = (unsigned short*)(ws + oW2L);
    float* decay  = (float*)(ws + oDec);
    float* invcnt = (float*)(ws + oInv);
    float* stbuf  = (float*)(ws + oSt);
    float* updbuf = (float*)(ws + oUpd);
    int*   bar    = (int*)(ws + oBar);

    k_prep_x   <<<4096, 256, 0, stream>>>(inputs, xbf);
    k_pack     <<<1024, 256, 0, stream>>>(W1, W2, W1L, W2L, p1, p2, p1L, p2L);
    k_prep_misc<<<128,  256, 0, stream>>>(decay_raw, decay, invcnt, stbuf, bar);
    k_scan     <<<MC,   256, 0, stream>>>(xbf, p1, p2, p1L, p2L,
                                          b1, b2, b1L, b2L,
                                          decay, invcnt, stbuf, updbuf,
                                          bar, bar + 1, out);
}
